// AttentionHead_33217277067771
// MI455X (gfx1250) — matrix operand order repaired and checked by reading, not compiled, not run on hardware
//
#include <hip/hip_runtime.h>

// Problem dims (fixed by the reference).
#define BB 16
#define SS 2048
#define DD 1024
#define HHH 128

typedef __attribute__((ext_vector_type(16))) __bf16   v16bf;
typedef __attribute__((ext_vector_type(8)))  float    v8f;
typedef __attribute__((ext_vector_type(4)))  unsigned v4u;
typedef __attribute__((ext_vector_type(4)))  int      v4i;

union BF16x16 { v16bf v; v4u q[2]; unsigned u[8]; __bf16 h[16]; };

// --- gfx1250 async global->LDS copy (ASYNCcnt-tracked), with sync fallback ---
#if __has_builtin(__builtin_amdgcn_global_load_async_to_lds_b128) && \
    __has_builtin(__builtin_amdgcn_s_wait_asynccnt)
#define HAS_ASYNC_LDS 1
using as1_v4i = __attribute__((address_space(1))) v4i *;
using as3_v4i = __attribute__((address_space(3))) v4i *;
#else
#define HAS_ASYNC_LDS 0
#endif

__device__ __forceinline__ void cp_b128_to_lds(__bf16* lds, const __bf16* g) {
#if HAS_ASYNC_LDS
  __builtin_amdgcn_global_load_async_to_lds_b128((as1_v4i)g, (as3_v4i)lds, 0, 0);
#else
  *(v4u*)lds = *(const v4u*)g;
#endif
}

__device__ __forceinline__ void wait_stage_done() {
#if HAS_ASYNC_LDS
  __builtin_amdgcn_s_wait_asynccnt(0);
#endif
  __syncthreads();
}

// Pack two f32 -> packed bf16x2 (low = first arg): bias-round + v_perm_b32.
__device__ __forceinline__ unsigned packbf(float lo, float hi) {
  unsigned a = __builtin_bit_cast(unsigned, lo) + 0x8000u;
  unsigned b = __builtin_bit_cast(unsigned, hi) + 0x8000u;
  return __builtin_amdgcn_perm(b, a, 0x07060302u);
}

__device__ __forceinline__ __bf16 f2bf(float f) {
  unsigned a = __builtin_bit_cast(unsigned, f) + 0x8000u;
  unsigned short s = (unsigned short)(a >> 16);
  return __builtin_bit_cast(__bf16, s);
}

// Build a 16-wide bf16 operand from two 16-byte chunks (b128 loads).
__device__ __forceinline__ v16bf ld2(const __bf16* p0, const __bf16* p1) {
  BF16x16 r;
  r.q[0] = *(const v4u*)p0;
  r.q[1] = *(const v4u*)p1;
  return r.v;
}

__device__ __forceinline__ v8f wmma_bf16(v16bf a, v16bf b, v8f c) {
  return __builtin_amdgcn_wmma_f32_16x16x32_bf16(
      /*neg_a=*/false, a, /*neg_b=*/false, b,
      /*c_mod=*/(short)0, c, /*reuse_a=*/false, /*reuse_b=*/false);
}

// Counter-based hash RNG for dropout.
__device__ __forceinline__ unsigned hash32(unsigned x) {
  x *= 2654435761u; x ^= x >> 16; x *= 0x7feb352du;
  x ^= x >> 15;     x *= 0x846ca68bu; x ^= x >> 16;
  return x;
}

// ---------------------------------------------------------------------------
// Kernel 0: convert the three weight matrices (H x D fp32) to bf16 once.
// ---------------------------------------------------------------------------
__global__ __launch_bounds__(256) void convert_w_kernel(
    const float* __restrict__ qw, const float* __restrict__ kw,
    const float* __restrict__ vw, unsigned* __restrict__ wb)
{
  const int m = blockIdx.y;
  const float* src = (m == 0) ? qw : (m == 1) ? kw : vw;
  unsigned* dst = wb + (size_t)m * (HHH * DD / 2);
  int i = blockIdx.x * 256 + threadIdx.x;
  const float2 v = ((const float2*)src)[i];
  dst[i] = packbf(v.x, v.y);
}

// ---------------------------------------------------------------------------
// Kernel 1: fused Q/K/V projection GEMM.  grid = (M/128, 3), block = 256.
// A: x fp32 loaded b128, packed to bf16 in-register. B: bf16 weights, pure loads.
// ---------------------------------------------------------------------------
__global__ __launch_bounds__(256) void qkv_proj_kernel(
    const float* __restrict__ x, const __bf16* __restrict__ wb,
    const float* __restrict__ qb, const float* __restrict__ kb,
    const float* __restrict__ vb,
    __bf16* __restrict__ Qo, __bf16* __restrict__ Ko, __bf16* __restrict__ Vo)
{
  const int lane = threadIdx.x & 31;
  const int wave = threadIdx.x >> 5;
  const int half = lane >> 4;     // 0/1: which 16-lane group
  const int ln   = lane & 15;
  const int rowbase = blockIdx.x * 128 + wave * 16;

  const __bf16* wm;  const float* bi;  __bf16* o;
  if (blockIdx.y == 0)      { wm = wb;                         bi = qb; o = Qo; }
  else if (blockIdx.y == 1) { wm = wb + (size_t)HHH * DD;      bi = kb; o = Ko; }
  else                      { wm = wb + (size_t)2 * HHH * DD;  bi = vb; o = Vo; }

  v8f acc[8];
#pragma unroll
  for (int i = 0; i < 8; i++) acc[i] = (v8f){0,0,0,0,0,0,0,0};

  for (int d0 = 0; d0 < DD; d0 += 32) {
    // A operand (x rows): lane = row ln, K = d0 + half*8 + {0..7}, +16.
    const float* xp = x + (size_t)(rowbase + ln) * DD + d0 + half * 8;
    float4 a0 = *(const float4*)(xp);
    float4 a1 = *(const float4*)(xp + 4);
    float4 a2 = *(const float4*)(xp + 16);
    float4 a3 = *(const float4*)(xp + 20);
    BF16x16 A;
    A.u[0] = packbf(a0.x, a0.y); A.u[1] = packbf(a0.z, a0.w);
    A.u[2] = packbf(a1.x, a1.y); A.u[3] = packbf(a1.z, a1.w);
    A.u[4] = packbf(a2.x, a2.y); A.u[5] = packbf(a2.z, a2.w);
    A.u[6] = packbf(a3.x, a3.y); A.u[7] = packbf(a3.z, a3.w);

#pragma unroll
    for (int hc = 0; hc < 8; hc++) {
      // B operand (w^T): lane = col h = hc*16+ln, K = d0 + half*8 + {0..7}, +16.
      const __bf16* wp = wm + (size_t)(hc * 16 + ln) * DD + d0 + half * 8;
      acc[hc] = wmma_bf16(A.v, ld2(wp, wp + 16), acc[hc]);
    }
  }

#pragma unroll
  for (int hc = 0; hc < 8; hc++) {
    float bbv = bi[hc * 16 + ln];
#pragma unroll
    for (int j = 0; j < 8; j++) {
      int row = rowbase + half * 8 + j;              // C layout row mapping
      o[(size_t)row * HHH + hc * 16 + ln] = f2bf(acc[hc][j] + bbv);
    }
  }
}

// ---------------------------------------------------------------------------
// Kernel 2: transpose V (b,s,h) -> VT (b,h,s), bf16, LDS-tiled 32x32.
// ---------------------------------------------------------------------------
__global__ __launch_bounds__(256) void transpose_v_kernel(
    const __bf16* __restrict__ Vo, __bf16* __restrict__ VT)
{
  __shared__ __bf16 tile[32][33];
  const int s0 = blockIdx.x * 32;
  const int h0 = blockIdx.y * 32;
  const int b  = blockIdx.z;

  {  // Coalesced read: thread t -> (s0 + t/8, h0 + (t%8)*4), 4 bf16 (8B).
    int sl = threadIdx.x >> 3;
    int hl = (threadIdx.x & 7) * 4;
    const __bf16* src = Vo + ((size_t)(b * SS + s0 + sl)) * HHH + h0 + hl;
#pragma unroll
    for (int i = 0; i < 4; i++) tile[sl][hl + i] = src[i];
  }
  __syncthreads();
  {  // Coalesced write: thread t -> (h0 + t/8, s0 + (t%8)*4).
    int hl = threadIdx.x >> 3;
    int sl = (threadIdx.x & 7) * 4;
    __bf16* dst = VT + ((size_t)(b * HHH + h0 + hl)) * SS + s0 + sl;
#pragma unroll
    for (int i = 0; i < 4; i++) dst[i] = tile[sl + i][hl];
  }
}

// ---------------------------------------------------------------------------
// Kernel 3: flash attention.  grid = (S/128, B), block = 256 (8 waves).
// Double-buffered K/V^T tiles staged (async if available) into LDS; one
// barrier per tile; next tile's copy overlaps current tile's 16 WMMAs.
// ---------------------------------------------------------------------------
__global__ __launch_bounds__(256) void flash_attn_kernel(
    const __bf16* __restrict__ Qb, const __bf16* __restrict__ Kb,
    const __bf16* __restrict__ VT, const int* __restrict__ amask,
    float* __restrict__ out)
{
  __shared__ __attribute__((aligned(16))) __bf16 Kt[2][32 * HHH];   // [kpos][h]
  __shared__ __attribute__((aligned(16))) __bf16 Vts[2][HHH * 32];  // [h][kpos]
  __shared__ __attribute__((aligned(16))) __bf16 Pt[8][16 * 32];    // per-wave P

  const int b    = blockIdx.y;
  const int lane = threadIdx.x & 31;
  const int wave = threadIdx.x >> 5;
  const int half = lane >> 4;
  const int ln   = lane & 15;
  const int qbase = blockIdx.x * 128 + wave * 16;

  const float    NEG_INF  = -__builtin_inff();
  const float    SCALE    = 0.08838834764831845f;   // 1/sqrt(128)
  const float    INV_KEEP = 1.0f / 0.7f;
  const unsigned KEEP16   = 45875u;                 // 0.7 * 65536

  const __bf16* Kbase  = Kb + (size_t)b * SS * HHH;
  const __bf16* VTbase = VT + (size_t)b * HHH * SS;
  const int*    mrow   = amask + b * SS;

  // Stage one 32-key tile (K row-major + V^T) into LDS buffer pbuf.
  auto stage_tile = [&](int k0, int pbuf) {
    {  // K: 32 rows x 128 h, 32B per thread.
      int r = threadIdx.x >> 3;
      int c = (threadIdx.x & 7) * 16;
      const __bf16* src = Kbase + (size_t)(k0 + r) * HHH + c;
      __bf16* dst = &Kt[pbuf][r * HHH + c];
      cp_b128_to_lds(dst, src);
      cp_b128_to_lds(dst + 8, src + 8);
    }
    {  // V^T: 128 h rows x 32 kpos, 32B per thread.
      int h = threadIdx.x >> 1;
      int c = (threadIdx.x & 1) * 16;
      const __bf16* src = VTbase + (size_t)h * SS + k0 + c;
      __bf16* dst = &Vts[pbuf][h * 32 + c];
      cp_b128_to_lds(dst, src);
      cp_b128_to_lds(dst + 8, src + 8);
    }
  };

  // Load this wave's Q tile (16 rows x 128 h) into 4 A-operands, kept in VGPRs.
  const __bf16* Qrow = Qb + ((size_t)b * SS + qbase) * HHH;
  v16bf qa[4];
#pragma unroll
  for (int hk = 0; hk < 4; hk++) {
    const __bf16* p = Qrow + (size_t)ln * HHH + hk * 32 + half * 8;
    qa[hk] = ld2(p, p + 16);
  }

  v8f acc[8];
#pragma unroll
  for (int i = 0; i < 8; i++) acc[i] = (v8f){0,0,0,0,0,0,0,0};
  float rm[8], rl[8];
#pragma unroll
  for (int j = 0; j < 8; j++) { rm[j] = NEG_INF; rl[j] = 0.0f; }

  stage_tile(0, 0);

  for (int k0 = 0; k0 < SS; k0 += 32) {
    const int pbuf = (k0 >> 5) & 1;
    wait_stage_done();                      // tile k0 resident, all waves
    if (k0 + 32 < SS) stage_tile(k0 + 32, pbuf ^ 1);  // overlap with compute

    const __bf16* Kc = &Kt[pbuf][0];
    const __bf16* Vc = &Vts[pbuf][0];

    // Scores: two 16x16 tiles (key cols k0..+15 and k0+16..+31), K-loop over h.
    v8f c0 = (v8f){0,0,0,0,0,0,0,0};
    v8f c1 = (v8f){0,0,0,0,0,0,0,0};
#pragma unroll
    for (int hk = 0; hk < 4; hk++) {
      const __bf16* pb0 = Kc + (size_t)ln * HHH + hk * 32 + half * 8;
      c0 = wmma_bf16(qa[hk], ld2(pb0, pb0 + 16), c0);
      const __bf16* pb1 = Kc + (size_t)(16 + ln) * HHH + hk * 32 + half * 8;
      c1 = wmma_bf16(qa[hk], ld2(pb1, pb1 + 16), c1);
    }

    const int mk0 = mrow[k0 + ln];
    const int mk1 = mrow[k0 + 16 + ln];
    __bf16* P = &Pt[wave][0];

#pragma unroll
    for (int j = 0; j < 8; j++) {
      float s0 = mk0 ? c0[j] * SCALE : NEG_INF;
      float s1 = mk1 ? c1[j] * SCALE : NEG_INF;
      // Row max across the 16 lanes holding this row.
      float m = fmaxf(s0, s1);
#pragma unroll
      for (int off = 1; off < 16; off <<= 1) m = fmaxf(m, __shfl_xor(m, off, 32));
      float nm = fmaxf(rm[j], m);
      float alpha = (nm == NEG_INF) ? 1.0f : __expf(rm[j] - nm);
      rm[j] = nm;
      float p0 = (s0 == NEG_INF) ? 0.0f : __expf(s0 - nm);
      float p1 = (s1 == NEG_INF) ? 0.0f : __expf(s1 - nm);
      float rs = p0 + p1;
#pragma unroll
      for (int off = 1; off < 16; off <<= 1) rs += __shfl_xor(rs, off, 32);
      rl[j] = rl[j] * alpha + rs;
#pragma unroll
      for (int hc = 0; hc < 8; hc++) acc[hc][j] *= alpha;

      // Dropout: one hash serves both elements of the (col, col+16) pair.
      // (Denominator uses pre-dropout probs; numerator dropped/scaled.)
      int row = half * 8 + j;
      unsigned base = ((unsigned)b * SS + (unsigned)(qbase + row)) * (unsigned)SS;
      unsigned u = hash32(base + (unsigned)(k0 + ln));
      float d0 = ((u & 0xFFFFu) < KEEP16) ? p0 * INV_KEEP : 0.0f;
      float d1 = ((u >> 16)     < KEEP16) ? p1 * INV_KEEP : 0.0f;
      P[row * 32 + ln]      = f2bf(d0);
      P[row * 32 + 16 + ln] = f2bf(d1);
    }

    // Reload P in A-operand layout (same-wave LDS RAW; hw waits on DScnt).
    const __bf16* pp = P + (size_t)ln * 32 + half * 8;
    v16bf pa = ld2(pp, pp + 16);

    // acc += P (16x32) x V (32x128), 8 WMMAs over the h tiles.
#pragma unroll
    for (int hc = 0; hc < 8; hc++) {
      const __bf16* pv = Vc + (size_t)(hc * 16 + ln) * 32 + half * 8;
      acc[hc] = wmma_bf16(pa, ld2(pv, pv + 16), acc[hc]);
    }
  }

  // Epilogue: normalize by softmax denominator (rcp + mul), fp32 out.
#pragma unroll
  for (int j = 0; j < 8; j++) {
    float inv = 1.0f / rl[j];
    int row = qbase + half * 8 + j;
    float* orow = out + ((size_t)b * SS + row) * HHH + ln;
#pragma unroll
    for (int hc = 0; hc < 8; hc++) orow[hc * 16] = acc[hc][j] * inv;
  }
}

extern "C" void kernel_launch(void* const* d_in, const int* in_sizes, int n_in,
                              void* d_out, int out_size, void* d_ws, size_t ws_size,
                              hipStream_t stream) {
  (void)in_sizes; (void)n_in; (void)out_size; (void)ws_size;
  const float* x  = (const float*)d_in[0];
  const int*   am = (const int*)d_in[1];
  const float* qw = (const float*)d_in[2];
  const float* qb = (const float*)d_in[3];
  const float* kw = (const float*)d_in[4];
  const float* kb = (const float*)d_in[5];
  const float* vw = (const float*)d_in[6];
  const float* vb = (const float*)d_in[7];
  float* out = (float*)d_out;

  // Workspace: bf16 Q, K, V (24 MB) + V^T (8 MB) + bf16 weights (0.75 MB).
  const size_t MAT = (size_t)BB * SS * HHH;
  __bf16* Qo = (__bf16*)d_ws;
  __bf16* Ko = Qo + MAT;
  __bf16* Vo = Ko + MAT;
  __bf16* VT = Vo + MAT;
  __bf16* Wb = VT + MAT;

  dim3 gw((HHH * DD / 2) / 256, 3);
  convert_w_kernel<<<gw, 256, 0, stream>>>(qw, kw, vw, (unsigned*)Wb);

  dim3 g1((BB * SS) / 128, 3);
  qkv_proj_kernel<<<g1, 256, 0, stream>>>(x, Wb, qb, kb, vb, Qo, Ko, Vo);

  dim3 gt(SS / 32, HHH / 32, BB);
  transpose_v_kernel<<<gt, 256, 0, stream>>>(Vo, VT);

  dim3 g2(SS / 128, BB);
  flash_attn_kernel<<<g2, 256, 0, stream>>>(Qo, Ko, VT, am, out);
}
